// GraphAttentionLayer_17179869618
// MI455X (gfx1250) — matrix operand order repaired and checked by reading, not compiled, not run on hardware
//
#include <hip/hip_runtime.h>
#include <hip/hip_bf16.h>
#include <stdint.h>

// ---------------------------------------------------------------------------
// GAT layer for MI455X (gfx1250, wave32, WMMA).
// B=8, N=1024, D=1024, H=8, F=128. Both GEMMs via v_wmma_f32_16x16x32_bf16.
// ---------------------------------------------------------------------------

typedef __attribute__((ext_vector_type(16))) __bf16 v16bf;
typedef __attribute__((ext_vector_type(8)))  __bf16 v8bf;
typedef __attribute__((ext_vector_type(8)))  float  v8f;

#define GAT_B   8
#define GAT_N   1024
#define GAT_D   1024
#define GAT_H   8
#define GAT_F   128
#define GAT_HF  1024   // H*F
#define GAT_BN  8192   // B*N

union Pack4 { uint2 u2; __bf16 b[4]; };

__device__ __forceinline__ v8f wmma_bf16(v16bf a, v16bf b, v8f c) {
    // (neg_a, A, neg_b, B, c_mod, C, reuse_a, reuse_b)
    return __builtin_amdgcn_wmma_f32_16x16x32_bf16(false, a, false, b,
                                                   (short)0, c, false, false);
}

// ---------------------------------------------------------------------------
// Kernel 1: g = h @ W   (8192x1024) x (1024x1024), output stored as bf16.
// Block tile 128 rows x 256 cols. 8 waves in 2x4 grid, each wave computes a
// 64x64 sub-tile = 4x4 WMMA accumulators (16 wmma per 16 ds_load_b128).
// K-step = 32. All LDS staging uses packed ds_store_b64.
// ---------------------------------------------------------------------------
__global__ __launch_bounds__(256) void k_gemm_g(const float* __restrict__ h,
                                                const float* __restrict__ W,
                                                __bf16* __restrict__ g) {
    __shared__ alignas(64) __bf16 ldsA[128][32];   // [row][k]
    __shared__ alignas(64) __bf16 ldsB[256][32];   // [col][k]  (W tile transposed)

    const int tid  = threadIdx.x;
    const int lane = tid & 31;
    const int wave = tid >> 5;
    const int wy   = wave >> 2;          // 0..1 -> 64-row group
    const int wx   = wave & 3;           // 0..3 -> 64-col group
    const int row0 = blockIdx.y * 128;   // over B*N
    const int col0 = blockIdx.x * 256;   // over H*F

    v8f acc[4][4] = {};

    const int m   = lane & 15;
    const int kbA = (lane >= 16) ? 8 : 0;
    const int kbB = (lane >= 16) ? 8 : 0;

    for (int k0 = 0; k0 < GAT_D; k0 += 32) {
        __syncthreads();
        // ---- stage A tile (h): float4 load -> packed ds_store_b64 ----
        for (int it = 0; it < 4; ++it) {
            int idx = it * 256 + tid;            // 0..1023 float4 segments
            int r   = idx >> 3;                  // 128 rows
            int sg  = idx & 7;                   // 8 float4 per row
            float4 v = *(const float4*)&h[(size_t)(row0 + r) * GAT_D + k0 + sg * 4];
            Pack4 p;
            p.b[0] = (__bf16)v.x; p.b[1] = (__bf16)v.y;
            p.b[2] = (__bf16)v.z; p.b[3] = (__bf16)v.w;
            *(uint2*)&ldsA[r][sg * 4] = p.u2;
        }
        // ---- stage B tile (W) transposed: column-major gather (coalesced in c,
        //      W is L2-resident) -> packed ds_store_b64 ----
        for (int it = 0; it < 8; ++it) {
            int idx = it * 256 + tid;            // 0..2047
            int c   = idx & 255;                 // 256 cols
            int kg  = idx >> 8;                  // 8 groups of 4 k
            Pack4 p;
            for (int q = 0; q < 4; ++q)
                p.b[q] = (__bf16)W[(size_t)(k0 + kg * 4 + q) * GAT_HF + col0 + c];
            *(uint2*)&ldsB[c][kg * 4] = p.u2;
        }
        __syncthreads();

        // ---- fragments per ISA VGPR layout ----
        v16bf a[4], b[4];
        for (int i = 0; i < 4; ++i) {
            int r = wy * 64 + i * 16 + m;
            ((v8bf*)&a[i])[0] = *(const v8bf*)&ldsA[r][kbA];        // K kbA..+7
            ((v8bf*)&a[i])[1] = *(const v8bf*)&ldsA[r][kbA + 16];   // K kbA+16..+23
        }
        for (int j = 0; j < 4; ++j) {
            int c = wx * 64 + j * 16 + m;                            // col n = lane&15
            ((v8bf*)&b[j])[0] = *(const v8bf*)&ldsB[c][kbB];         // K kbB..+7
            ((v8bf*)&b[j])[1] = *(const v8bf*)&ldsB[c][kbB + 16];    // K kbB+16..+23
        }
        for (int i = 0; i < 4; ++i)
            for (int j = 0; j < 4; ++j)
                acc[i][j] = wmma_bf16(a[i], b[j], acc[i][j]);
    }

    // ---- store C tiles (bf16 g) per C layout: VGPR v -> M = v + 8*(lane>=16) ----
    const int n   = lane & 15;
    const int mro = (lane >= 16) ? 8 : 0;
    for (int i = 0; i < 4; ++i)
        for (int j = 0; j < 4; ++j) {
            int gr0 = row0 + wy * 64 + i * 16 + mro;
            int gc  = col0 + wx * 64 + j * 16 + n;
            for (int v = 0; v < 8; ++v)
                g[(size_t)(gr0 + v) * GAT_HF + gc] = (__bf16)acc[i][j][v];
        }
}

// ---------------------------------------------------------------------------
// Kernel 2: el[b,n,h] = sum_f g*a_l[f] ; er likewise. One wave per (b,n,h).
// ---------------------------------------------------------------------------
__global__ __launch_bounds__(256) void k_elr(const __bf16* __restrict__ g,
                                             const float* __restrict__ a,
                                             float* __restrict__ el,
                                             float* __restrict__ er) {
    const int wid  = blockIdx.x * 8 + (threadIdx.x >> 5); // (b*n)*H + h
    const int lane = threadIdx.x & 31;
    const int bn   = wid >> 3;
    const int hh   = wid & 7;
    float sl = 0.f, sr = 0.f;
    for (int t = 0; t < 4; ++t) {
        int   f  = t * 32 + lane;
        float gv = (float)g[(size_t)bn * GAT_HF + hh * GAT_F + f];
        sl += gv * a[f];
        sr += gv * a[GAT_F + f];
    }
    for (int o = 16; o; o >>= 1) {
        sl += __shfl_xor(sl, o, 32);
        sr += __shfl_xor(sr, o, 32);
    }
    if (lane == 0) { el[wid] = sl; er[wid] = sr; }
}

// ---------------------------------------------------------------------------
// Kernel 3: per (b,i,h): row max m and sum s of exp(e-m) over j (online).
// One wave per (b,i); all 8 heads in registers. Self-loop: j==i unmasked.
// ---------------------------------------------------------------------------
__global__ __launch_bounds__(256) void k_ms(const int*   __restrict__ adj,
                                            const float* __restrict__ el,
                                            const float* __restrict__ er,
                                            float* __restrict__ mout,
                                            float* __restrict__ sout) {
    const int wid  = blockIdx.x * 8 + (threadIdx.x >> 5); // b*N + i
    const int lane = threadIdx.x & 31;
    const int b    = wid >> 10;
    const int i    = wid & 1023;

    float elv[GAT_H], m[GAT_H], s[GAT_H];
    for (int hh = 0; hh < GAT_H; ++hh) {
        elv[hh] = el[(size_t)wid * GAT_H + hh];
        m[hh] = -3.0e38f; s[hh] = 0.f;
    }
    const int* arow = adj + (size_t)wid * GAT_N;
    for (int t = 0; t < 32; ++t) {
        int  j  = t * 32 + lane;
        bool ok = (arow[j] != 0) || (j == i);
        const float* erp = &er[((size_t)(b << 10) + j) * GAT_H];
        float4 e0 = *(const float4*)erp;
        float4 e1 = *(const float4*)(erp + 4);
        float erv[GAT_H] = {e0.x, e0.y, e0.z, e0.w, e1.x, e1.y, e1.z, e1.w};
        for (int hh = 0; hh < GAT_H; ++hh) {
            float e = elv[hh] + erv[hh];
            e = (e >= 0.f) ? e : 0.2f * e;
            if (!ok) e = -1000.f;
            if (e > m[hh]) { s[hh] = s[hh] * __expf(m[hh] - e) + 1.f; m[hh] = e; }
            else           { s[hh] += __expf(e - m[hh]); }
        }
    }
    for (int o = 16; o; o >>= 1)
        for (int hh = 0; hh < GAT_H; ++hh) {
            float om = __shfl_xor(m[hh], o, 32);
            float os = __shfl_xor(s[hh], o, 32);
            float nm = fmaxf(m[hh], om);
            s[hh] = s[hh] * __expf(m[hh] - nm) + os * __expf(om - nm);
            m[hh] = nm;
        }
    if (lane == 0)
        for (int hh = 0; hh < GAT_H; ++hh) {
            mout[(size_t)wid * GAT_H + hh] = m[hh];
            sout[(size_t)wid * GAT_H + hh] = s[hh];
        }
}

// ---------------------------------------------------------------------------
// Kernel 4: out[b,i,h,f] = sum_j P[i,j] * g[j,h,f]  via WMMA.
// Block = (64 i rows) x (all 128 f) for one (b,h). 8 waves -> 16 f each,
// 4 accumulators (i 0-15,16-31,32-47,48-63). K-step = 32 j. P tile built in
// LDS (bf16); g tile staged transposed [f][j] with packed ds_store_b64.
// ---------------------------------------------------------------------------
__global__ __launch_bounds__(256) void k_attn(const __bf16* __restrict__ g,
                                              const int*   __restrict__ adj,
                                              const float* __restrict__ el,
                                              const float* __restrict__ er,
                                              const float* __restrict__ mrow,
                                              const float* __restrict__ srow,
                                              float* __restrict__ out) {
    __shared__ alignas(64) __bf16 ldsP[64][32];    // [i][j]
    __shared__ alignas(64) __bf16 ldsG[128][32];   // [f][j]  (g tile transposed)
    __shared__ float elS[64], mS[64], invS[64];

    const int tid  = threadIdx.x;
    const int lane = tid & 31;
    const int wave = tid >> 5;
    const int i0   = blockIdx.x * 64;
    const int hh   = blockIdx.y;
    const int b    = blockIdx.z;
    const int fb   = wave * 16;

    if (tid < 64) {
        size_t gi = ((size_t)(b << 10) + i0 + tid) * GAT_H + hh;
        elS[tid]  = el[gi];
        mS[tid]   = mrow[gi];
        invS[tid] = 1.0f / srow[gi];
    }

    v8f acc[4] = {};
    const int m   = lane & 15;
    const int kbA = (lane >= 16) ? 8 : 0;
    const int kbB = (lane >= 16) ? 8 : 0;

    for (int j0 = 0; j0 < GAT_N; j0 += 32) {
        __syncthreads();
        // ---- build normalized probability tile P (bf16): 64x32 ----
        for (int t = 0; t < 8; ++t) {
            int idx = t * 256 + tid;         // 0..2047
            int il  = idx >> 5, jl = idx & 31;
            int i   = i0 + il,  j  = j0 + jl;
            float e = elS[il] + er[((size_t)(b << 10) + j) * GAT_H + hh];
            e = (e >= 0.f) ? e : 0.2f * e;
            bool ok = (adj[((size_t)(b << 10) + i) * GAT_N + j] != 0) || (j == i);
            if (!ok) e = -1000.f;
            float p = __expf(e - mS[il]) * invS[il];
            ldsP[il][jl] = (__bf16)p;
        }
        // ---- stage g tile transposed: ldsG[f][j], packed ds_store_b64 ----
        for (int t = 0; t < 4; ++t) {
            int idx = t * 256 + tid;         // 0..1023
            int f   = idx & 127;             // coalesced over f
            int jg  = idx >> 7;              // 8 groups of 4 j
            Pack4 p;
            for (int q = 0; q < 4; ++q)
                p.b[q] = g[((size_t)((b << 10) + j0 + jg * 4 + q)) * GAT_HF
                           + hh * GAT_F + f];
            *(uint2*)&ldsG[f][jg * 4] = p.u2;
        }
        __syncthreads();

        v16bf a[4], bf;
        for (int i = 0; i < 4; ++i) {
            int r = i * 16 + m;
            ((v8bf*)&a[i])[0] = *(const v8bf*)&ldsP[r][kbA];
            ((v8bf*)&a[i])[1] = *(const v8bf*)&ldsP[r][kbA + 16];
        }
        ((v8bf*)&bf)[0] = *(const v8bf*)&ldsG[fb + m][kbB];
        ((v8bf*)&bf)[1] = *(const v8bf*)&ldsG[fb + m][kbB + 16];

        for (int i = 0; i < 4; ++i)
            acc[i] = wmma_bf16(a[i], bf, acc[i]);
    }

    // ---- store out (f32) per C layout ----
    const int n = lane & 15;
    const int r = (lane >= 16) ? 8 : 0;
    for (int i = 0; i < 4; ++i)
        for (int v = 0; v < 8; ++v)
            out[((size_t)((b << 10) + i0 + i * 16 + r + v)) * GAT_HF
                + hh * GAT_F + fb + n] = acc[i][v];
}

// ---------------------------------------------------------------------------
// Host launcher
// ---------------------------------------------------------------------------
extern "C" void kernel_launch(void* const* d_in, const int* in_sizes, int n_in,
                              void* d_out, int out_size, void* d_ws, size_t ws_size,
                              hipStream_t stream) {
    const float* h   = (const float*)d_in[0];
    const int*   adj = (const int*)d_in[1];
    const float* W   = (const float*)d_in[2];
    const float* a   = (const float*)d_in[3];
    float*       out = (float*)d_out;

    // Workspace layout
    char* ws = (char*)d_ws;
    __bf16* g_bf16 = (__bf16*)ws;                                   // 16 MB
    size_t off = (size_t)GAT_BN * GAT_HF * sizeof(__bf16);
    float* el = (float*)(ws + off);  off += (size_t)GAT_BN * GAT_H * sizeof(float);
    float* er = (float*)(ws + off);  off += (size_t)GAT_BN * GAT_H * sizeof(float);
    float* mr = (float*)(ws + off);  off += (size_t)GAT_BN * GAT_H * sizeof(float);
    float* sr = (float*)(ws + off);

    // 1) g = h @ W (bf16 WMMA)
    k_gemm_g<<<dim3(GAT_HF / 256, GAT_BN / 128), 256, 0, stream>>>(h, W, g_bf16);
    // 2) el / er
    k_elr<<<(GAT_BN * GAT_H) / 8, 256, 0, stream>>>(g_bf16, a, el, er);
    // 3) softmax row max & sum
    k_ms<<<GAT_BN / 8, 256, 0, stream>>>(adj, el, er, mr, sr);
    // 4) out = softmax(e) @ g (bf16 WMMA)
    k_attn<<<dim3(GAT_N / 64, GAT_H, GAT_B), 256, 0, stream>>>(g_bf16, adj, el, er, mr, sr, out);
}
